// Model_86517821216165
// MI455X (gfx1250) — matrix operand order repaired and checked by reading, not compiled, not run on hardware
//
#include <hip/hip_runtime.h>
#include <hip/hip_bf16.h>
#include <math.h>

#define BB 4
#define TQ 64
#define DMODEL 256
#define NH 4
#define DHEAD 64
#define DFFC 1024
#define NLAYER 2
#define NMELC 80
#define TMELC 63

typedef __attribute__((ext_vector_type(16))) _Float16 v16h;
typedef __attribute__((ext_vector_type(8)))  float    v8f;

#define WMMA_F16(a,b,c) __builtin_amdgcn_wmma_f32_16x16x32_f16(false,(a),false,(b),(short)0,(c),false,false)

// ---------------- WMMA fragment helpers (ISA 7.12.2 layouts, wave32) ----------------
// A (16x32 f16, M x K): lanes 0-15 -> row M=lane, K in {0..7, 16..23}; lanes 16-31 -> K in {8..15, 24..31}
__device__ __forceinline__ v16h load_frag_a(const _Float16* p, int ld) {
    int lane = threadIdx.x & 31;
    int row  = lane & 15;
    int k0   = (lane >> 4) << 3;            // 0 or 8
    const uint4* q0 = (const uint4*)(p + row * ld + k0);
    const uint4* q1 = (const uint4*)(p + row * ld + k0 + 16);
    v16h r;
    ((uint4*)&r)[0] = *q0;
    ((uint4*)&r)[1] = *q1;
    return r;
}
// B (32x16 f16, K x N) sourced from a pre-transposed N x K buffer:
// lanes 0-15 -> col N=lane, K in {0..7, 16..23}; lanes 16-31 -> K in {8..15, 24..31}
__device__ __forceinline__ v16h load_frag_b(const _Float16* p, int ld) {
    int lane = threadIdx.x & 31;
    int n    = lane & 15;
    int k0   = (lane >> 4) << 3;            // 0 or 8
    const uint4* q = (const uint4*)(p + n * ld + k0);
    v16h r;
    ((uint4*)&r)[0] = q[0];
    ((uint4*)&r)[1] = q[2];
    return r;
}
__device__ __forceinline__ float wred_max(float v) {
    for (int o = 16; o; o >>= 1) v = fmaxf(v, __shfl_xor(v, o, 32));
    return v;
}
__device__ __forceinline__ float wred_sum(float v) {
    for (int o = 16; o; o >>= 1) v += __shfl_xor(v, o, 32);
    return v;
}
__device__ __forceinline__ v8f vzero8() {
    v8f z = {0.f,0.f,0.f,0.f,0.f,0.f,0.f,0.f};
    return z;
}

// ---------------- one-time prep kernels ----------------
__device__ __forceinline__ float pe_val(int t, int c) {
    double ang = (double)t / pow(10000.0, (double)(2 * (c / 2)) / 256.0);
    return (float)((c & 1) ? cos(ang) : sin(ang));
}

__global__ void k_convert_wT(const float* __restrict__ W, _Float16* __restrict__ WT, int K, int N) {
    int idx = blockIdx.x * blockDim.x + threadIdx.x;
    if (idx >= K * N) return;
    int k = idx / N, n = idx % N;
    WT[(size_t)n * K + k] = (_Float16)W[idx];
}

__global__ void k_prep_enc0(const int* __restrict__ x, const float* __restrict__ tok, float* __restrict__ ench) {
    int idx = blockIdx.x * blockDim.x + threadIdx.x;
    if (idx >= BB * TQ * DMODEL) return;
    int c = idx & 255, t = (idx >> 8) & 63, b = idx >> 14;
    ench[idx] = tok[x[b * TQ + t] * DMODEL + c] * 16.0f + pe_val(t, c);
}

__global__ void k_prep_spk(const int* __restrict__ ids, const float* __restrict__ emb,
                           const float* __restrict__ w, const float* __restrict__ bb,
                           float* __restrict__ spk) {
    int idx = blockIdx.x * blockDim.x + threadIdx.x;
    if (idx >= BB * DMODEL) return;
    int c = idx & 255, b = idx >> 8;
    const float* e = emb + ids[b] * 64;
    float s = bb[c];
    for (int k = 0; k < 64; ++k) s += e[k] * w[k * DMODEL + c];
    spk[idx] = tanhf(s);
}

__global__ void k_prep_decin(const float* __restrict__ y,
                             const float* __restrict__ w1, const float* __restrict__ b1,
                             const float* __restrict__ w2, const float* __restrict__ b2,
                             const float* __restrict__ spk, float* __restrict__ din) {
    int b = blockIdx.x >> 6, t = blockIdx.x & 63, c = threadIdx.x;
    __shared__ float hid[256];
    if (t == 0) { din[(b * TQ) * DMODEL + c] = spk[b * DMODEL + c] + pe_val(0, c); return; }
    const float* yr = y + ((size_t)b * TMELC + (t - 1)) * NMELC;
    float s = b1[c];
    for (int k = 0; k < NMELC; ++k) s += yr[k] * w1[k * 256 + c];
    hid[c] = fmaxf(s, 0.f);
    __syncthreads();
    float o = b2[c];
    for (int k = 0; k < 256; ++k) o += hid[k] * w2[k * DMODEL + c];
    din[(b * TQ + t) * DMODEL + c] = fmaxf(o, 0.f) + pe_val(t, c);
}

__global__ void k_init_center(float* __restrict__ center) {
    if (threadIdx.x < BB) center[threadIdx.x] = 1.0f;
}

// ---------------- fused MHA (+residual+LN, optional center update) ----------------
// biasMode: 0=none (enc self), 1=causal (dec self), 2=gaussian (cross)
__global__ __launch_bounds__(512) void k_mha(
    const float* __restrict__ hq, const float* __restrict__ kvmain,
    const float* __restrict__ spk, int Tk,
    const _Float16* __restrict__ wqT, const _Float16* __restrict__ wkT,
    const _Float16* __restrict__ wvT, const _Float16* __restrict__ woT,
    const float* __restrict__ lng, const float* __restrict__ lnb,
    int biasMode, float* __restrict__ center, int qi, int writeCenter,
    float* __restrict__ outp)
{
    __shared__ __align__(16) _Float16 s_x [80 * 264];   // staged input (hq then kv), reused for preLN
    __shared__ __align__(16) _Float16 s_Q [64 * 264];   // Q, all heads
    __shared__ __align__(16) _Float16 s_O [64 * 264];   // attention output, all heads
    __shared__ __align__(16) _Float16 s_K [80 * 72];    // per-head K (key-major)
    __shared__ __align__(16) _Float16 s_VT[64 * 104];   // per-head V^T (feat-major)
    __shared__ __align__(16) _Float16 s_P [64 * 104];   // per-head softmax probs (f16)
    __shared__ __align__(16) float    s_S [64 * 84];    // per-head scores (f32)
    __shared__ float s_cent[96];

    const int b    = blockIdx.x;
    const int tid  = threadIdx.x;
    const int wid  = tid >> 5;
    const int lane = tid & 31;
    const int nlm  = lane & 15;     // D-frag N coordinate
    const int msel = lane >> 4;     // D-frag M half-select

    const int Tkpad = (Tk > 64) ? 80 : 64;
    const int PK    = (Tk > 64) ? 96 : 64;
    const int KT    = Tkpad >> 4;

    float c0 = (biasMode == 2) ? center[b] : 0.f;
    if (tid < 96) s_cent[tid] = 0.f;

    // ---- phase 1: stage hq (f32->f16), Q = hq @ Wq ----
    for (int idx = tid; idx < 64 * 256; idx += 512) {
        int r = idx >> 8, c = idx & 255;
        s_x[r * 264 + c] = (_Float16)hq[((size_t)b * TQ + r) * DMODEL + c];
    }
    __syncthreads();
    for (int t = wid; t < 64; t += 16) {
        int mt = t & 3, nt = t >> 2;
        v8f acc = vzero8();
        for (int kk = 0; kk < 256; kk += 32)
            acc = WMMA_F16(load_frag_a(s_x + (mt * 16) * 264 + kk, 264),
                           load_frag_b(wqT + (size_t)(nt * 16) * 256 + kk, 256), acc);
        for (int r = 0; r < 8; ++r)
            s_Q[(mt * 16 + msel * 8 + r) * 264 + nt * 16 + nlm] = (_Float16)acc[r];
    }
    __syncthreads();

    // ---- phase 2: stage kv (optionally prepend spk; zero pad keys) ----
    for (int idx = tid; idx < Tkpad * 256; idx += 512) {
        int r = idx >> 8, c = idx & 255;
        float v;
        if (spk) {
            if (r == 0)      v = spk[b * DMODEL + c];
            else if (r < Tk) v = kvmain[((size_t)b * TQ + (r - 1)) * DMODEL + c];
            else             v = 0.f;
        } else {
            v = (r < Tk) ? kvmain[((size_t)b * TQ + r) * DMODEL + c] : 0.f;
        }
        s_x[r * 264 + c] = (_Float16)v;
    }
    if (PK > Tkpad) { // zero V^T pad columns so P(0) * garbage never appears
        int padw = PK - Tkpad;
        for (int idx = tid; idx < 64 * padw; idx += 512) {
            int r = idx / padw, c = idx % padw;
            s_VT[r * 104 + Tkpad + c] = (_Float16)0.f;
        }
    }
    __syncthreads();

    for (int hd = 0; hd < NH; ++hd) {
        // K/V projections for this head
        for (int t = wid; t < KT * 4; t += 16) {
            int mt = t % KT, nt = t / KT;
            v8f accK = vzero8(), accV = vzero8();
            for (int kk = 0; kk < 256; kk += 32) {
                v16h a = load_frag_a(s_x + (mt * 16) * 264 + kk, 264);
                accK = WMMA_F16(a, load_frag_b(wkT + (size_t)(hd * 64 + nt * 16) * 256 + kk, 256), accK);
                accV = WMMA_F16(a, load_frag_b(wvT + (size_t)(hd * 64 + nt * 16) * 256 + kk, 256), accV);
            }
            for (int r = 0; r < 8; ++r) {
                int key = mt * 16 + msel * 8 + r;
                int ft  = nt * 16 + nlm;
                s_K[key * 72 + ft]  = (_Float16)accK[r];
                s_VT[ft * 104 + key] = (_Float16)accV[r];
            }
        }
        __syncthreads();
        // scores S = Q_h K_h^T / 8 + bias
        for (int t = wid; t < 4 * KT; t += 16) {
            int mt = t % 4, nt = t / 4;
            v8f acc = vzero8();
            for (int kk = 0; kk < 64; kk += 32)
                acc = WMMA_F16(load_frag_a(s_Q + (mt * 16) * 264 + hd * 64 + kk, 264),
                               load_frag_b(s_K + (nt * 16) * 72 + kk, 72), acc);
            for (int r = 0; r < 8; ++r) {
                int q = mt * 16 + msel * 8 + r;
                int k = nt * 16 + nlm;
                float s = acc[r] * 0.125f;
                if (biasMode == 1)      { if (k > q) s = -1e9f; }
                else if (biasMode == 2) { float dk = (float)k - c0; s += -0.005f * dk * dk; }
                if (k >= Tk) s = -1e9f;
                s_S[q * 84 + k] = s;
            }
        }
        __syncthreads();
        // softmax (rows statically mapped to waves -> race-free center accumulation)
        for (int r = wid; r < 64; r += 16) {
            float mx = -1e30f;
            for (int c = lane; c < Tkpad; c += 32) mx = fmaxf(mx, s_S[r * 84 + c]);
            mx = wred_max(mx);
            float sum = 0.f;
            for (int c = lane; c < Tkpad; c += 32) sum += __expf(s_S[r * 84 + c] - mx);
            sum = wred_sum(sum);
            float inv = 1.f / sum;
            for (int c = lane; c < PK; c += 32) {
                float p = (c < Tkpad) ? __expf(s_S[r * 84 + c] - mx) * inv : 0.f;
                s_P[r * 104 + c] = (_Float16)p;
                if (writeCenter && r == qi && c < Tk) s_cent[c] += p * 0.25f;  // mean over 4 heads
            }
        }
        __syncthreads();
        // O_h = P V_h
        {
            int mt = wid & 3, nt = wid >> 2;
            v8f acc = vzero8();
            for (int kk = 0; kk < PK; kk += 32)
                acc = WMMA_F16(load_frag_a(s_P + (mt * 16) * 104 + kk, 104),
                               load_frag_b(s_VT + (nt * 16) * 104 + kk, 104), acc);
            for (int r = 0; r < 8; ++r)
                s_O[(mt * 16 + msel * 8 + r) * 264 + hd * 64 + nt * 16 + nlm] = (_Float16)acc[r];
        }
        __syncthreads();
    }

    if (writeCenter && tid == 0) {
        float cc = 0.f;
        for (int k = 0; k < Tk; ++k) cc += s_cent[k] * (float)k;
        center[b] = cc;
    }

    // ---- phase 3: out = O @ Wo + residual, then LN ----
    for (int t = wid; t < 64; t += 16) {
        int mt = t & 3, nt = t >> 2;
        v8f acc = vzero8();
        for (int kk = 0; kk < 256; kk += 32)
            acc = WMMA_F16(load_frag_a(s_O + (mt * 16) * 264 + kk, 264),
                           load_frag_b(woT + (size_t)(nt * 16) * 256 + kk, 256), acc);
        for (int r = 0; r < 8; ++r) {
            int m = mt * 16 + msel * 8 + r;
            int n = nt * 16 + nlm;
            float v = acc[r] + hq[((size_t)b * TQ + m) * DMODEL + n];
            s_x[m * 264 + n] = (_Float16)v;      // s_x reused as pre-LN buffer
        }
    }
    __syncthreads();
    for (int r = wid; r < 64; r += 16) {
        float sum = 0.f;
        for (int c = lane; c < 256; c += 32) sum += (float)s_x[r * 264 + c];
        sum = wred_sum(sum);
        float mu = sum * (1.f / 256.f);
        float vs = 0.f;
        for (int c = lane; c < 256; c += 32) { float d = (float)s_x[r * 264 + c] - mu; vs += d * d; }
        vs = wred_sum(vs);
        float rstd = rsqrtf(vs * (1.f / 256.f) + 1e-5f);
        for (int c = lane; c < 256; c += 32) {
            float d = ((float)s_x[r * 264 + c] - mu) * rstd;
            outp[((size_t)b * TQ + r) * DMODEL + c] = d * lng[c] + lnb[c];
        }
    }
}

// ---------------- fused FFN (+residual+LN, in place) ----------------
__global__ __launch_bounds__(512) void k_ffn(
    float* __restrict__ h,
    const _Float16* __restrict__ w1T, const float* __restrict__ b1,
    const _Float16* __restrict__ w2T, const float* __restrict__ b2,
    const float* __restrict__ lng, const float* __restrict__ lnb)
{
    __shared__ __align__(16) _Float16 s_x  [64 * 264];
    __shared__ __align__(16) _Float16 s_mid[64 * 264];
    const int b = blockIdx.x, tid = threadIdx.x, wid = tid >> 5, lane = tid & 31;
    const int nlm = lane & 15, msel = lane >> 4;

    for (int idx = tid; idx < 64 * 256; idx += 512) {
        int r = idx >> 8, c = idx & 255;
        s_x[r * 264 + c] = (_Float16)h[((size_t)b * TQ + r) * DMODEL + c];
    }
    __syncthreads();

    v8f acc0 = vzero8(), acc1 = vzero8(), acc2 = vzero8(), acc3 = vzero8();
    for (int ch = 0; ch < 4; ++ch) {
        // mid chunk = relu(x @ W1[:, ch*256..] + b1)
        for (int t = wid; t < 64; t += 16) {
            int mt = t & 3, nt = t >> 2;
            v8f a = vzero8();
            for (int kk = 0; kk < 256; kk += 32)
                a = WMMA_F16(load_frag_a(s_x + (mt * 16) * 264 + kk, 264),
                             load_frag_b(w1T + (size_t)(ch * 256 + nt * 16) * 256 + kk, 256), a);
            for (int r = 0; r < 8; ++r) {
                int m = mt * 16 + msel * 8 + r, n = nt * 16 + nlm;
                float v = a[r] + b1[ch * 256 + n];
                s_mid[m * 264 + n] = (_Float16)fmaxf(v, 0.f);
            }
        }
        __syncthreads();
        // accumulate into this wave's fixed output column block n0 = wid*16
        const _Float16* bsrc = w2T + (size_t)(wid * 16) * DFFC + ch * 256;
        for (int kk = 0; kk < 256; kk += 32) {
            v16h bf = load_frag_b(bsrc + kk, DFFC);
            acc0 = WMMA_F16(load_frag_a(s_mid + 0  * 264 + kk, 264), bf, acc0);
            acc1 = WMMA_F16(load_frag_a(s_mid + 16 * 264 + kk, 264), bf, acc1);
            acc2 = WMMA_F16(load_frag_a(s_mid + 32 * 264 + kk, 264), bf, acc2);
            acc3 = WMMA_F16(load_frag_a(s_mid + 48 * 264 + kk, 264), bf, acc3);
        }
        __syncthreads();
    }
    // epilogue: + b2 + residual -> preLN (reuse s_mid)
    {
        int n = wid * 16 + nlm;
        float bb2 = b2[n];
        for (int r = 0; r < 8; ++r) {
            int m0 = msel * 8 + r;
            s_mid[(m0 +  0) * 264 + n] = (_Float16)(acc0[r] + bb2 + h[((size_t)b * TQ + m0 +  0) * DMODEL + n]);
            s_mid[(m0 + 16) * 264 + n] = (_Float16)(acc1[r] + bb2 + h[((size_t)b * TQ + m0 + 16) * DMODEL + n]);
            s_mid[(m0 + 32) * 264 + n] = (_Float16)(acc2[r] + bb2 + h[((size_t)b * TQ + m0 + 32) * DMODEL + n]);
            s_mid[(m0 + 48) * 264 + n] = (_Float16)(acc3[r] + bb2 + h[((size_t)b * TQ + m0 + 48) * DMODEL + n]);
        }
    }
    __syncthreads();
    for (int r = wid; r < 64; r += 16) {
        float sum = 0.f;
        for (int c = lane; c < 256; c += 32) sum += (float)s_mid[r * 264 + c];
        sum = wred_sum(sum);
        float mu = sum * (1.f / 256.f);
        float vs = 0.f;
        for (int c = lane; c < 256; c += 32) { float d = (float)s_mid[r * 264 + c] - mu; vs += d * d; }
        vs = wred_sum(vs);
        float rstd = rsqrtf(vs * (1.f / 256.f) + 1e-5f);
        for (int c = lane; c < 256; c += 32) {
            float d = ((float)s_mid[r * 264 + c] - mu) * rstd;
            h[((size_t)b * TQ + r) * DMODEL + c] = d * lng[c] + lnb[c];
        }
    }
}

// ---------------- per-step output head ----------------
__global__ void k_head(const float* __restrict__ dec_h, int qi,
                       const float* __restrict__ mel_w, const float* __restrict__ mel_b,
                       const float* __restrict__ stop_w, const float* __restrict__ stop_b,
                       float* __restrict__ outp) {
    int b = blockIdx.x;
    __shared__ float row[256];
    for (int c = threadIdx.x; c < 256; c += blockDim.x)
        row[c] = dec_h[((size_t)b * TQ + qi) * DMODEL + c];
    __syncthreads();
    for (int j = threadIdx.x; j < NMELC; j += blockDim.x) {
        float s = mel_b[j];
        for (int c = 0; c < 256; ++c) s += row[c] * mel_w[c * NMELC + j];
        outp[((size_t)b * TMELC + qi) * NMELC + j] = s;
    }
    if (threadIdx.x == 0) {
        float s = stop_b[0];
        for (int c = 0; c < 256; ++c) s += row[c] * stop_w[c];
        outp[(size_t)BB * TMELC * NMELC + b * TMELC + qi] = 1.f / (1.f + __expf(-s));
    }
}

// ---------------- host orchestration ----------------
extern "C" void kernel_launch(void* const* d_in, const int* in_sizes, int n_in,
                              void* d_out, int out_size, void* d_ws, size_t ws_size,
                              hipStream_t stream) {
    (void)in_sizes; (void)n_in; (void)out_size; (void)ws_size;
    const int*   x        = (const int*)  d_in[0];
    const int*   speakers = (const int*)  d_in[1];
    const float* y        = (const float*)d_in[2];
    const float* tok_emb  = (const float*)d_in[4];
    const float* spk_emb  = (const float*)d_in[5];
    const float* spk_w    = (const float*)d_in[6];
    const float* spk_b    = (const float*)d_in[7];
    const float* pre_w1   = (const float*)d_in[8];
    const float* pre_b1   = (const float*)d_in[9];
    const float* pre_w2   = (const float*)d_in[10];
    const float* pre_b2   = (const float*)d_in[11];
    const float* enc_qkvo = (const float*)d_in[12];
    const float* enc_ln   = (const float*)d_in[13];
    const float* enc_f1   = (const float*)d_in[14];
    const float* enc_fb1  = (const float*)d_in[15];
    const float* enc_f2   = (const float*)d_in[16];
    const float* enc_fb2  = (const float*)d_in[17];
    const float* dself    = (const float*)d_in[18];
    const float* dcross   = (const float*)d_in[19];
    const float* dec_ln   = (const float*)d_in[20];
    const float* dec_f1   = (const float*)d_in[21];
    const float* dec_fb1  = (const float*)d_in[22];
    const float* dec_f2   = (const float*)d_in[23];
    const float* dec_fb2  = (const float*)d_in[24];
    const float* mel_w    = (const float*)d_in[25];
    const float* mel_b    = (const float*)d_in[26];
    const float* stop_w   = (const float*)d_in[27];
    const float* stop_b   = (const float*)d_in[28];
    float* outp = (float*)d_out;

    char* ws = (char*)d_ws;
    size_t off = 0;
    auto take = [&](size_t bytes) -> char* {
        char* p = ws + off;
        off = (off + bytes + 255) & ~(size_t)255;
        return p;
    };
    _Float16* wEnc   = (_Float16*)take((size_t)NLAYER * 4 * 65536 * 2);
    _Float16* wSelf  = (_Float16*)take((size_t)NLAYER * 4 * 65536 * 2);
    _Float16* wCross = (_Float16*)take((size_t)NLAYER * 4 * 65536 * 2);
    _Float16* wEF1   = (_Float16*)take((size_t)NLAYER * 262144 * 2);
    _Float16* wEF2   = (_Float16*)take((size_t)NLAYER * 262144 * 2);
    _Float16* wDF1   = (_Float16*)take((size_t)NLAYER * 262144 * 2);
    _Float16* wDF2   = (_Float16*)take((size_t)NLAYER * 262144 * 2);
    float* enc_h  = (float*)take((size_t)BB * TQ * DMODEL * 4);
    float* dec_h  = (float*)take((size_t)BB * TQ * DMODEL * 4);
    float* dec_in = (float*)take((size_t)BB * TQ * DMODEL * 4);
    float* spk    = (float*)take((size_t)BB * DMODEL * 4);
    float* center = (float*)take(64);

    // one-time: transpose+convert all GEMM weights to f16 (N x K layout)
    for (int l = 0; l < NLAYER; ++l) {
        for (int j = 0; j < 4; ++j) {
            size_t o = (size_t)(l * 4 + j) * 65536;
            k_convert_wT<<<256, 256, 0, stream>>>(enc_qkvo + o, wEnc + o, 256, 256);
            k_convert_wT<<<256, 256, 0, stream>>>(dself    + o, wSelf + o, 256, 256);
            k_convert_wT<<<256, 256, 0, stream>>>(dcross   + o, wCross + o, 256, 256);
        }
        size_t of = (size_t)l * 262144;
        k_convert_wT<<<1024, 256, 0, stream>>>(enc_f1 + of, wEF1 + of, 256, 1024);
        k_convert_wT<<<1024, 256, 0, stream>>>(enc_f2 + of, wEF2 + of, 1024, 256);
        k_convert_wT<<<1024, 256, 0, stream>>>(dec_f1 + of, wDF1 + of, 256, 1024);
        k_convert_wT<<<1024, 256, 0, stream>>>(dec_f2 + of, wDF2 + of, 1024, 256);
    }
    k_prep_enc0<<<(BB * TQ * DMODEL + 255) / 256, 256, 0, stream>>>(x, tok_emb, enc_h);
    k_prep_spk<<<(BB * DMODEL + 255) / 256, 256, 0, stream>>>(speakers, spk_emb, spk_w, spk_b, spk);
    k_prep_decin<<<BB * TQ, 256, 0, stream>>>(y, pre_w1, pre_b1, pre_w2, pre_b2, spk, dec_in);
    k_init_center<<<1, 32, 0, stream>>>(center);

    for (int i = 1; i <= TMELC; ++i) {
        int qi = i - 1;
        for (int l = 0; l < NLAYER; ++l) {
            const _Float16* e = wEnc + (size_t)(l * 4) * 65536;
            k_mha<<<BB, 512, 0, stream>>>(enc_h, enc_h, (const float*)nullptr, 64,
                e, e + 65536, e + 2 * 65536, e + 3 * 65536,
                enc_ln + ((l * 2 + 0) * 2 + 0) * 256, enc_ln + ((l * 2 + 0) * 2 + 1) * 256,
                0, (float*)nullptr, qi, 0, enc_h);
            k_ffn<<<BB, 512, 0, stream>>>(enc_h,
                wEF1 + (size_t)l * 262144, enc_fb1 + l * 1024,
                wEF2 + (size_t)l * 262144, enc_fb2 + l * 256,
                enc_ln + ((l * 2 + 1) * 2 + 0) * 256, enc_ln + ((l * 2 + 1) * 2 + 1) * 256);

            const float* hin = (l == 0) ? dec_in : dec_h;
            const _Float16* s = wSelf + (size_t)(l * 4) * 65536;
            k_mha<<<BB, 512, 0, stream>>>(hin, hin, (const float*)nullptr, 64,
                s, s + 65536, s + 2 * 65536, s + 3 * 65536,
                dec_ln + ((l * 3 + 0) * 2 + 0) * 256, dec_ln + ((l * 3 + 0) * 2 + 1) * 256,
                1, (float*)nullptr, qi, 0, dec_h);

            const _Float16* cx = wCross + (size_t)(l * 4) * 65536;
            k_mha<<<BB, 512, 0, stream>>>(dec_h, enc_h, spk, 65,
                cx, cx + 65536, cx + 2 * 65536, cx + 3 * 65536,
                dec_ln + ((l * 3 + 1) * 2 + 0) * 256, dec_ln + ((l * 3 + 1) * 2 + 1) * 256,
                2, center, qi, (l == NLAYER - 1) ? 1 : 0, dec_h);

            k_ffn<<<BB, 512, 0, stream>>>(dec_h,
                wDF1 + (size_t)l * 262144, dec_fb1 + l * 1024,
                wDF2 + (size_t)l * 262144, dec_fb2 + l * 256,
                dec_ln + ((l * 3 + 2) * 2 + 0) * 256, dec_ln + ((l * 3 + 2) * 2 + 1) * 256);
        }
        k_head<<<BB, 128, 0, stream>>>(dec_h, qi, mel_w, mel_b, stop_w, stop_b, outp);
    }
}
